// CGCNN_62251255989043
// MI455X (gfx1250) — matrix operand order repaired and checked by reading, not compiled, not run on hardware
//
#include <hip/hip_runtime.h>
#include <hip/hip_bf16.h>

// ---------------- problem constants (match reference) ----------------
constexpr int N   = 20000;
constexpr int E   = 500000;
constexpr int G   = 64;
constexpr int NF  = 64;     // node feature dim
constexpr int NG  = 100;    // gaussians
constexpr int LC  = 6;      // conv layers
constexpr int FC  = 128;
constexpr int NFC = 3;
constexpr int EMB = 92;
constexpr float EPS = 1e-5f;

constexpr int KPAD   = 256;           // 228 padded to 8 k-chunks of 32
constexpr int ZC     = 2 * NF;        // 128 z columns
constexpr int NTILES = E / 16;        // 31250 M-tiles of 16 edges

// ---------------- workspace layout (bytes, 256-aligned) ----------------
constexpr size_t OFF_X32  = 0;                                   // N*64 f32
constexpr size_t OFF_X16  = OFF_X32 + (size_t)N * 64 * 4;        // N*64 f16
constexpr size_t OFF_EF16 = OFF_X16 + (size_t)N * 64 * 2;        // E*128 f16
constexpr size_t OFF_Z16  = OFF_EF16 + (size_t)E * 128 * 2;      // E*128 f16
constexpr size_t OFF_AGG  = OFF_Z16 + (size_t)E * 128 * 2;       // N*64 f32
constexpr size_t OFF_WPK  = OFF_AGG + (size_t)N * 64 * 4;        // 6*8*8*32*16 f16
constexpr size_t OFF_BNS  = OFF_WPK + (size_t)LC * 8 * 8 * 32 * 16 * 2; // 256 f32 (sum,sumsq)
constexpr size_t OFF_SCL  = OFF_BNS + 1024;                      // 256 f32 (scale,shift)
constexpr size_t OFF_MOL  = OFF_SCL + 1024;                      // G*64 f32
constexpr size_t OFF_CNT  = OFF_MOL + (size_t)G * 64 * 4;        // G f32

typedef __attribute__((ext_vector_type(16))) _Float16 v16h;
typedef __attribute__((ext_vector_type(8)))  float    v8f;

union Frag16 { v16h v; uint4 q[2]; _Float16 h[16]; };

__device__ __forceinline__ float softplusf(float x) {
    return fmaxf(x, 0.0f) + log1pf(__expf(-fabsf(x)));
}
__device__ __forceinline__ float sigmoidf(float x) {
    return 1.0f / (1.0f + __expf(-x));
}

// ---------------- generic zero ----------------
__global__ void zero_kernel(float* __restrict__ p, long long n) {
    long long i = (long long)blockIdx.x * blockDim.x + threadIdx.x;
    if (i < n) p[i] = 0.0f;
}

// ---------------- pack conv_W into pre-swizzled WMMA B-fragments ----------------
// Wpack[l][nt(8)][kt(8)][lane(32)][j(16)], lane's 16 halves contiguous (32B).
// B[k][n]: lane L -> n = nt*16 + (L&15); element j -> k = kt*32 + ((L>>4)<<4) + j.
__global__ void pack_w_kernel(const float* __restrict__ convW, _Float16* __restrict__ wpack) {
    int idx = blockIdx.x * blockDim.x + threadIdx.x;
    const int total = LC * 8 * 8 * 32 * 16;
    if (idx >= total) return;
    int j    = idx & 15;
    int lane = (idx >> 4) & 31;
    int kt   = (idx >> 9) & 7;
    int nt   = (idx >> 12) & 7;
    int l    = idx >> 15;
    int n = nt * 16 + (lane & 15);
    int k = kt * 32 + ((lane >> 4) << 4) + j;
    float w = (k < 2 * NF + NG) ? convW[((size_t)l * (2 * NF + NG) + k) * ZC + n] : 0.0f;
    wpack[idx] = (_Float16)w;
}

// ---------------- gaussian smearing -> ef16 [E,128] (cols 100..127 = 0) ----------------
__global__ void ef_kernel(const float* __restrict__ dist, _Float16* __restrict__ ef16) {
    long long idx = (long long)blockIdx.x * blockDim.x + threadIdx.x;
    if (idx >= (long long)E * 128) return;
    int e = (int)(idx >> 7);
    int g = (int)(idx & 127);
    float v = 0.0f;
    if (g < NG) {
        float d = dist[e];
        float off = (float)g * (6.0f / 99.0f);
        float t = d - off;
        v = __expf(-136.125f * t * t);   // -0.5/(6/99)^2
    }
    ef16[idx] = (_Float16)v;
}

// ---------------- node embedding: x = emb[an] @ nucW + nucb ----------------
__global__ void embed_kernel(const int* __restrict__ an, const float* __restrict__ emb,
                             const float* __restrict__ nucW, const float* __restrict__ nucb,
                             float* __restrict__ x32, _Float16* __restrict__ x16) {
    int idx = blockIdx.x * blockDim.x + threadIdx.x;
    if (idx >= N * NF) return;
    int n = idx >> 6, f = idx & 63;
    int a = an[n];
    float acc = nucb[f];
    const float* er = emb + (size_t)a * EMB;
    for (int k = 0; k < EMB; ++k) acc = fmaf(er[k], nucW[k * NF + f], acc);
    x32[idx] = acc;
    x16[idx] = (_Float16)acc;
}

// ---------------- edge GEMM: z = [x[dst]|x[src]|ef] @ W  (WMMA f16->f32) ----------------
// block = 256 threads = 8 waves; wave w owns N-tile w (16 cols). BN column stats fused.
// A-tile (16 edges x 256 K halves = 8KB) double-buffered via async global->LDS DMA:
// tile t+stride streams into buf^1 while WMMAs consume buf (ASYNCcnt in-order per wave).
__global__ void edge_gemm_kernel(const _Float16* __restrict__ x16,
                                 const _Float16* __restrict__ ef16,
                                 const _Float16* __restrict__ wpack,   // this layer's pack
                                 const int* __restrict__ nbr,          // [2,E]: src then dst
                                 _Float16* __restrict__ z16,
                                 float* __restrict__ bnsum,            // [128]
                                 float* __restrict__ bnsq) {           // [128]
    __shared__ __align__(16) _Float16 sA[2][16 * KPAD];   // 2 x 8 KB A tiles
    const int tid  = threadIdx.x;
    const int wv   = tid >> 5;        // wave id == N-tile id
    const int lane = tid & 31;

    // loop-invariant B fragments for this wave: 8 k-chunks, 32B contiguous per lane
    v16h bfr[8];
#pragma unroll
    for (int kt = 0; kt < 8; ++kt) {
        Frag16 u;
        const uint4* p = (const uint4*)(wpack + (((size_t)wv * 8 + kt) * 32 + lane) * 16);
        u.q[0] = p[0];
        u.q[1] = p[1];
        bfr[kt] = u.v;
    }

    const int row  = lane & 15;            // A-row / C column select
    const int hi   = lane >> 4;            // half-wave id
    const int base = hi << 4;              // A frag k-offset (0 or 16)
    const int col  = wv * 16 + row;        // z column for this lane

    const unsigned sAbase = (unsigned)(uintptr_t)(&sA[0][0]);   // LDS byte offset

    // issue async A-tile gather for tile t into buffer buf (2 x b128 per thread)
    auto issue_tile = [&](int t, int buf) {
        const int e0 = t * 16;
#pragma unroll
        for (int half = 0; half < 2; ++half) {
            int i   = tid + half * 256;        // segment id 0..511
            int r   = i >> 5;                  // edge row within tile
            int seg = i & 31;                  // 16B segment within row
            unsigned long long gaddr;
            if (seg < 8) {
                int d = nbr[E + e0 + r];                               // dst gather
                gaddr = (unsigned long long)(uintptr_t)(x16 + (size_t)d * NF + seg * 8);
            } else if (seg < 16) {
                int s = nbr[e0 + r];                                   // src gather
                gaddr = (unsigned long long)(uintptr_t)(x16 + (size_t)s * NF + (seg - 8) * 8);
            } else {
                gaddr = (unsigned long long)(uintptr_t)(ef16 + (size_t)(e0 + r) * 128 + (seg - 16) * 8);
            }
            unsigned ldsoff = sAbase + (unsigned)buf * 8192u + (unsigned)i * 16u;
            asm volatile("global_load_async_to_lds_b128 %0, %1, off"
                         :: "v"(ldsoff), "v"(gaddr) : "memory");
        }
    };

    float lsum = 0.0f, lsq = 0.0f;

    int buf = 0;
    if ((int)blockIdx.x < NTILES) issue_tile(blockIdx.x, 0);

    for (int t = blockIdx.x; t < NTILES; t += gridDim.x) {
        const int tn = t + (int)gridDim.x;
        if (tn < NTILES) {
            // previous iteration's trailing barrier guarantees buf^1 is free
            issue_tile(tn, buf ^ 1);
            __builtin_prefetch(ef16 + (size_t)tn * 16 * 128, 0, 1);
            // allow the 2 just-issued loads to stay outstanding; older pair must land
            asm volatile("s_wait_asynccnt 0x2" ::: "memory");
        } else {
            asm volatile("s_wait_asynccnt 0x0" ::: "memory");
        }
        __syncthreads();                    // all waves' current-tile DMA visible

        const int e0 = t * 16;
        v8f c = {0.f, 0.f, 0.f, 0.f, 0.f, 0.f, 0.f, 0.f};
        const _Float16* arow = &sA[buf][row * KPAD];
#pragma unroll
        for (int kt = 0; kt < 8; ++kt) {
            Frag16 a;
            a.q[0] = *(const uint4*)(arow + kt * 32 + base);
            a.q[1] = *(const uint4*)(arow + kt * 32 + base + 8);
            c = __builtin_amdgcn_wmma_f32_16x16x32_f16(
                    false, a.v, false, bfr[kt], (short)0, c, false, false);
        }

        // epilogue: one base address, eight b16 stores at immediate offsets r*ZC
        _Float16* zptr = z16 + (size_t)(e0 + (hi << 3)) * ZC + col;
#pragma unroll
        for (int r = 0; r < 8; ++r) {
            float v = c[r];
            zptr[r * ZC] = (_Float16)v;
            lsum += v;
            lsq  = fmaf(v, v, lsq);
        }
        __syncthreads();                    // all reads of sA[buf] done before reuse
        buf ^= 1;
    }

    // combine the two half-waves (same column, different rows), then one atomic each
    lsum += __shfl_xor(lsum, 16, 32);
    lsq  += __shfl_xor(lsq, 16, 32);
    if (hi == 0) {
        atomicAdd(&bnsum[col], lsum);
        atomicAdd(&bnsq[col], lsq);
    }
}

// ---------------- BN finalize -> per-column scale/shift (conv_b cancels) ----------------
__global__ void bn_finalize_kernel(const float* __restrict__ bnsum, const float* __restrict__ bnsq,
                                   const float* __restrict__ bng, const float* __restrict__ bnb,
                                   float* __restrict__ scale, float* __restrict__ shift) {
    int t = threadIdx.x;
    if (t >= ZC) return;
    float mu  = bnsum[t] * (1.0f / (float)E);
    float var = bnsq[t] * (1.0f / (float)E) - mu * mu;
    float rs  = rsqrtf(var + EPS);
    float g   = bng[t] * rs;
    scale[t] = g;
    shift[t] = bnb[t] - mu * g;
}

// ---------------- gated message + scatter-add to agg (2 features / thread) ----------------
__global__ void message_kernel(const _Float16* __restrict__ z16,
                               const float* __restrict__ scale, const float* __restrict__ shift,
                               const int* __restrict__ nbr, float* __restrict__ agg) {
    long long idx = (long long)blockIdx.x * blockDim.x + threadIdx.x;
    if (idx >= (long long)E * 32) return;
    int e = (int)(idx >> 5);
    int f = (int)((idx & 31) << 1);
    union { unsigned u; _Float16 h[2]; } za, zb;
    // z is read exactly once per layer: non-temporal so it doesn't evict ef16 from L2
    za.u = __builtin_nontemporal_load((const unsigned*)(z16 + (size_t)e * ZC + f));
    zb.u = __builtin_nontemporal_load((const unsigned*)(z16 + (size_t)e * ZC + NF + f));
    float z10 = fmaf((float)za.h[0], scale[f],          shift[f]);
    float z11 = fmaf((float)za.h[1], scale[f + 1],      shift[f + 1]);
    float z20 = fmaf((float)zb.h[0], scale[NF + f],     shift[NF + f]);
    float z21 = fmaf((float)zb.h[1], scale[NF + f + 1], shift[NF + f + 1]);
    float m0 = sigmoidf(z10) * softplusf(z20);
    float m1 = sigmoidf(z11) * softplusf(z21);
    int d = nbr[E + e];
    atomicAdd(&agg[(size_t)d * NF + f],     m0);
    atomicAdd(&agg[(size_t)d * NF + f + 1], m1);
}

// ---------------- per-node LayerNorm + residual softplus (wave32 per node) ----------------
__global__ void node_update_kernel(const float* __restrict__ agg,
                                   const float* __restrict__ lng, const float* __restrict__ lnb,
                                   float* __restrict__ x32, _Float16* __restrict__ x16) {
    int n  = blockIdx.x * 8 + (threadIdx.x >> 5);
    int ln = threadIdx.x & 31;
    if (n >= N) return;
    float a0 = agg[(size_t)n * NF + ln];
    float a1 = agg[(size_t)n * NF + 32 + ln];
    float s = a0 + a1;
#pragma unroll
    for (int off = 16; off >= 1; off >>= 1) s += __shfl_xor(s, off, 32);
    float mu = s * (1.0f / NF);
    float d0 = a0 - mu, d1 = a1 - mu;
    float q = d0 * d0 + d1 * d1;
#pragma unroll
    for (int off = 16; off >= 1; off >>= 1) q += __shfl_xor(q, off, 32);
    float rs = rsqrtf(q * (1.0f / NF) + EPS);
    float h0 = fmaf(d0 * rs, lng[ln],      lnb[ln]);
    float h1 = fmaf(d1 * rs, lng[32 + ln], lnb[32 + ln]);
    float nx0 = softplusf(h0 + x32[(size_t)n * NF + ln]);
    float nx1 = softplusf(h1 + x32[(size_t)n * NF + 32 + ln]);
    x32[(size_t)n * NF + ln]      = nx0;
    x32[(size_t)n * NF + 32 + ln] = nx1;
    x16[(size_t)n * NF + ln]      = (_Float16)nx0;
    x16[(size_t)n * NF + 32 + ln] = (_Float16)nx1;
}

// ---------------- mean pool per graph ----------------
__global__ void pool_kernel(const float* __restrict__ x32, const int* __restrict__ batch,
                            float* __restrict__ mol, float* __restrict__ cnt) {
    int idx = blockIdx.x * blockDim.x + threadIdx.x;
    if (idx >= N * NF) return;
    int n = idx >> 6, f = idx & 63;
    int b = batch[n];
    atomicAdd(&mol[(size_t)b * NF + f], x32[idx]);
    if (f == 0) atomicAdd(&cnt[b], 1.0f);
}

// ---------------- MLP head (single block, 128 threads) ----------------
__global__ void head_kernel(const float* __restrict__ mol, const float* __restrict__ cnt,
                            const float* __restrict__ fc1W, const float* __restrict__ fc1b,
                            const float* __restrict__ fcsW, const float* __restrict__ fcsb,
                            const float* __restrict__ outW, const float* __restrict__ outb,
                            float* __restrict__ out) {
    __shared__ float smol[G * NF];      // 16 KB
    __shared__ float sh[G * FC];        // 32 KB
    int t = threadIdx.x;
    for (int i = t; i < G * NF; i += FC)
        smol[i] = mol[i] / fmaxf(cnt[i >> 6], 1.0f);
    __syncthreads();

    for (int g = 0; g < G; ++g) {
        float acc = fc1b[t];
        for (int k = 0; k < NF; ++k) acc = fmaf(smol[g * NF + k], fc1W[k * FC + t], acc);
        sh[g * FC + t] = softplusf(acc);
    }
    __syncthreads();

    float r[G];
    for (int i = 0; i < NFC; ++i) {
        for (int g = 0; g < G; ++g) {
            float acc = fcsb[i * FC + t];
            const float* wcol = fcsW + (size_t)i * FC * FC;
            for (int k = 0; k < FC; ++k) acc = fmaf(sh[g * FC + k], wcol[k * FC + t], acc);
            r[g] = softplusf(acc);
        }
        __syncthreads();
        for (int g = 0; g < G; ++g) sh[g * FC + t] = r[g];
        __syncthreads();
    }

    if (t < G) {
        float acc = outb[0];
        for (int k = 0; k < FC; ++k) acc = fmaf(sh[t * FC + k], outW[k], acc);
        out[t] = acc;
    }
}

// ---------------- launcher ----------------
extern "C" void kernel_launch(void* const* d_in, const int* in_sizes, int n_in,
                              void* d_out, int out_size, void* d_ws, size_t ws_size,
                              hipStream_t stream) {
    (void)in_sizes; (void)n_in; (void)out_size; (void)ws_size;
    const int*   an    = (const int*)d_in[0];
    const int*   nbr   = (const int*)d_in[1];
    const float* dist  = (const float*)d_in[2];
    const int*   batch = (const int*)d_in[3];
    const float* emb   = (const float*)d_in[4];
    const float* nucW  = (const float*)d_in[5];
    const float* nucb  = (const float*)d_in[6];
    const float* convW = (const float*)d_in[7];
    // d_in[8] = conv_b: unused — cancels exactly inside BatchNorm
    const float* bng   = (const float*)d_in[9];
    const float* bnb   = (const float*)d_in[10];
    const float* lng   = (const float*)d_in[11];
    const float* lnb   = (const float*)d_in[12];
    const float* fc1W  = (const float*)d_in[13];
    const float* fc1b  = (const float*)d_in[14];
    const float* fcsW  = (const float*)d_in[15];
    const float* fcsb  = (const float*)d_in[16];
    const float* outW  = (const float*)d_in[17];
    const float* outb  = (const float*)d_in[18];

    char* ws = (char*)d_ws;
    float*    x32   = (float*)   (ws + OFF_X32);
    _Float16* x16   = (_Float16*)(ws + OFF_X16);
    _Float16* ef16  = (_Float16*)(ws + OFF_EF16);
    _Float16* z16   = (_Float16*)(ws + OFF_Z16);
    float*    agg   = (float*)   (ws + OFF_AGG);
    _Float16* wpack = (_Float16*)(ws + OFF_WPK);
    float*    bns   = (float*)   (ws + OFF_BNS);   // [0:128]=sum, [128:256]=sumsq
    float*    scl   = (float*)   (ws + OFF_SCL);   // [0:128]=scale, [128:256]=shift
    float*    mol   = (float*)   (ws + OFF_MOL);
    float*    cnt   = (float*)   (ws + OFF_CNT);
    float*    y     = (float*)d_out;

    // one-time prep (deterministic; re-run every call)
    pack_w_kernel<<<(LC * 8 * 8 * 32 * 16 + 255) / 256, 256, 0, stream>>>(convW, wpack);
    ef_kernel<<<(int)(((long long)E * 128 + 255) / 256), 256, 0, stream>>>(dist, ef16);
    embed_kernel<<<(N * NF + 255) / 256, 256, 0, stream>>>(an, emb, nucW, nucb, x32, x16);

    for (int l = 0; l < LC; ++l) {
        zero_kernel<<<((long long)N * NF + 255) / 256, 256, 0, stream>>>(agg, (long long)N * NF);
        zero_kernel<<<1, 256, 0, stream>>>(bns, 256);
        edge_gemm_kernel<<<1024, 256, 0, stream>>>(
            x16, ef16, wpack + (size_t)l * 8 * 8 * 32 * 16, nbr, z16, bns, bns + ZC);
        bn_finalize_kernel<<<1, ZC, 0, stream>>>(bns, bns + ZC, bng + l * ZC, bnb + l * ZC,
                                                 scl, scl + ZC);
        message_kernel<<<(int)(((long long)E * 32 + 255) / 256), 256, 0, stream>>>(
            z16, scl, scl + ZC, nbr, agg);
        node_update_kernel<<<(N + 7) / 8, 256, 0, stream>>>(agg, lng + l * NF, lnb + l * NF,
                                                            x32, x16);
    }

    zero_kernel<<<(G * NF + 255) / 256, 256, 0, stream>>>(mol, G * NF);
    zero_kernel<<<1, 256, 0, stream>>>(cnt, G);
    pool_kernel<<<(N * NF + 255) / 256, 256, 0, stream>>>(x32, batch, mol, cnt);
    head_kernel<<<1, FC, 0, stream>>>(mol, cnt, fc1W, fc1b, fcsW, fcsb, outW, outb, y);
}
